// BQQLinearInference_82635170775233
// MI455X (gfx1250) — matrix operand order repaired and checked by reading, not compiled, not run on hardware
//
#include <hip/hip_runtime.h>
#include <hip/hip_bf16.h>

// BQQ inference folded to: out = act_scale * (Xq_int16f · W^T) + bias
// Xq: integer-valued f16 (exact), W: folded f16 weights (built per call in ws).
// Main GEMM is HBM-bound (~44MB @ 23.3TB/s ≈ 2us); WMMA f16 w/ f32 acc.
// Tiles staged to LDS by the Tensor Data Mover with D# LDS padding so that
// row pitch = 144B (36 banks) -> conflict-free ds_load fragment feeds.

typedef __attribute__((ext_vector_type(16))) _Float16 v16h;
typedef __attribute__((ext_vector_type(8)))  _Float16 v8h;
typedef __attribute__((ext_vector_type(8)))  float    v8f;
typedef __attribute__((ext_vector_type(4)))  unsigned int u32x4;
typedef __attribute__((ext_vector_type(8)))  int      i32x8;
typedef __attribute__((ext_vector_type(4)))  int      i32x4;

#define PP 2
#define JJ 32
#define KK 32
#define MM 128
#define LL 16
#define NN 128
#define CIN  (KK*NN)   // 4096
#define COUT (JJ*MM)   // 4096
#define QMAX 127.0f

#define KSTEP 64              // k elements per LDS stage (2 WMMAs deep)
#define PITCH 72              // LDS row pitch in halves: 64 data + 8 pad (144B)

// ---------------- Kernel 1: per-block max/min partials ----------------
__global__ __launch_bounds__(256)
void k_minmax_partial(const float* __restrict__ x, int n,
                      float* __restrict__ pmax, float* __restrict__ pmin) {
  __shared__ float smx[256], smn[256];
  float mx = -INFINITY, mn = INFINITY;
  for (int i = blockIdx.x * blockDim.x + threadIdx.x; i < n;
       i += gridDim.x * blockDim.x) {
    float v = x[i];
    mx = fmaxf(mx, v);
    mn = fminf(mn, v);
  }
  smx[threadIdx.x] = mx;
  smn[threadIdx.x] = mn;
  __syncthreads();
  for (int off = 128; off > 0; off >>= 1) {
    if (threadIdx.x < off) {
      smx[threadIdx.x] = fmaxf(smx[threadIdx.x], smx[threadIdx.x + off]);
      smn[threadIdx.x] = fminf(smn[threadIdx.x], smn[threadIdx.x + off]);
    }
    __syncthreads();
  }
  if (threadIdx.x == 0) {
    pmax[blockIdx.x] = smx[0];
    pmin[blockIdx.x] = smn[0];
  }
}

// ---------------- Kernel 2: finalize act_scale ----------------
__global__ __launch_bounds__(256)
void k_scale(const float* __restrict__ pmax, const float* __restrict__ pmin,
             float* __restrict__ scalep) {
  __shared__ float smx[256], smn[256];
  int t = threadIdx.x;
  smx[t] = pmax[t];
  smn[t] = pmin[t];
  __syncthreads();
  for (int off = 128; off > 0; off >>= 1) {
    if (t < off) {
      smx[t] = fmaxf(smx[t], smx[t + off]);
      smn[t] = fminf(smn[t], smn[t + off]);
    }
    __syncthreads();
  }
  if (t == 0)
    scalep[0] = fmaxf((smx[0] - smn[0]) * (1.0f / (2.0f * QMAX)), 1e-8f);
}

// ---------------- Kernel 3: quantize X -> integer-valued f16 ----------------
__global__ __launch_bounds__(256)
void k_quant(const float* __restrict__ x, const float* __restrict__ scalep,
             _Float16* __restrict__ xq, int n) {
  const float inv = 1.0f / scalep[0];
  for (int i = blockIdx.x * blockDim.x + threadIdx.x; i < n;
       i += gridDim.x * blockDim.x) {
    float q = rintf(x[i] * inv);            // round-to-nearest-even, like jnp.round
    q = fminf(fmaxf(q, -QMAX), QMAX);
    xq[i] = (_Float16)q;                    // |q| <= 127 -> exact in f16
  }
}

// ---------------- Kernel 4: fold all four terms into W (f16) ----------------
__global__ __launch_bounds__(256)
void k_buildW(const float* __restrict__ Ysign, const float* __restrict__ Zsign,
              const float* __restrict__ Yscale, const float* __restrict__ Zscale,
              const float* __restrict__ A, _Float16* __restrict__ W) {
  __shared__ float Ys[PP * MM * LL];  // [p][m*16+l]
  __shared__ float Zs[PP * LL * NN];  // [p][l*128+n]
  const int j = blockIdx.x >> 5;
  const int k = blockIdx.x & 31;
  const int tid = threadIdx.x;

  float c0[PP], b1[PP], c2[PP];
  float dconst = 0.0f;
#pragma unroll
  for (int p = 0; p < PP; ++p) {
    const int base = (p * JJ + j) * KK + k;
    const float ysc = Yscale[base];
    const float zsc = Zscale[base];
    const float* a = A + (size_t)base * 4;
    c0[p] = a[0] * ysc * zsc;
    b1[p] = a[1] * ysc;
    c2[p] = a[2] * zsc;
    dconst += a[3];

    const float4* ysrc = (const float4*)(Ysign + (size_t)base * (MM * LL));
    const float4* zsrc = (const float4*)(Zsign + (size_t)base * (LL * NN));
    float4* yd = (float4*)&Ys[p * MM * LL];
    float4* zd = (float4*)&Zs[p * LL * NN];
    yd[tid] = ysrc[tid];
    yd[tid + 256] = ysrc[tid + 256];
    zd[tid] = zsrc[tid];
    zd[tid + 256] = zsrc[tid + 256];
  }
  __syncthreads();

  for (int idx = tid; idx < MM * NN; idx += 256) {
    const int m = idx >> 7;
    const int n = idx & 127;
    float acc = dconst;
#pragma unroll
    for (int p = 0; p < PP; ++p) {
      const float* yr = &Ys[p * MM * LL + m * LL];
      const float* zc = &Zs[p * LL * NN + n];
      float s = 0.0f, sy = 0.0f, sz = 0.0f;
#pragma unroll
      for (int l = 0; l < LL; ++l) {
        const float ys = yr[l];
        const float zs = zc[l * NN];
        s  = fmaf(ys, zs, s);
        sy += ys;
        sz += zs;
      }
      acc += c0[p] * s + b1[p] * sy + c2[p] * sz;
    }
    W[(size_t)(j * MM + m) * CIN + k * NN + n] = (_Float16)acc;
  }
}

// ---------------- TDM helper: 2-D tile load (row-major) into LDS ----------------
// D# per CDNA5 ISA §8.3/8.4. tensor_dim = remaining extent from the tile origin
// (OOB reads return zero -> free edge clamping). data_size = 2 bytes (f16).
// LDS padding: 16B pad after every 128B row -> 144B pitch (bank-conflict-free).
__device__ __forceinline__ void tdm_load_tile_f16(
    unsigned lds_byte_addr, const _Float16* gptr,
    unsigned tile_w, unsigned tile_h,
    unsigned rem_w, unsigned rem_h, unsigned row_stride_elems) {
  const unsigned long long ga = (unsigned long long)(uintptr_t)gptr;
  u32x4 g0;
  g0[0] = 1u;                                   // count=1 valid descriptor
  g0[1] = lds_byte_addr;                        // LDS dest (byte address)
  g0[2] = (unsigned)(ga & 0xFFFFFFFFu);         // global_addr[31:0]
  g0[3] = (unsigned)((ga >> 32) & 0x01FFFFFFu)  // global_addr[56:32]
          | (2u << 30);                         // type=2 ("image")
  i32x8 g1;
  g1[0] = (int)((1u << 16)                      // data_size=1 (2B); mask=0
                | (1u << 20)                    // pad_enable
                | (4u << 22)                    // pad_interval: 32 DWORDs (128B)
                | (3u << 25));                  // pad_amount: 4 DWORDs (16B)
  g1[1] = (int)((rem_w & 0xFFFFu) << 16);       // tensor_dim0[15:0]
  g1[2] = (int)(((rem_w >> 16) & 0xFFFFu) | ((rem_h & 0xFFFFu) << 16));
  g1[3] = (int)(((rem_h >> 16) & 0xFFFFu) | (tile_w << 16));   // tile_dim0
  g1[4] = (int)(tile_h & 0xFFFFu);              // tile_dim1 (tile_dim2=0)
  g1[5] = (int)row_stride_elems;                // tensor_dim0_stride[31:0]
  g1[6] = 0;
  g1[7] = 0;
  i32x4 z4 = {0, 0, 0, 0};
#if __clang_major__ >= 23
  i32x8 z8 = {0, 0, 0, 0, 0, 0, 0, 0};
  __builtin_amdgcn_tensor_load_to_lds(g0, g1, z4, z4, z8, 0);
#else
  __builtin_amdgcn_tensor_load_to_lds(g0, g1, z4, z4, 0);
#endif
}

__device__ __forceinline__ unsigned lds_off(const void* p) {
  // flat LDS addresses carry the LDS byte offset in the low 32 bits
  return (unsigned)(unsigned long long)(uintptr_t)p;
}

__device__ __forceinline__ v16h cat8(v8h lo, v8h hi) {
  return __builtin_shufflevector(lo, hi, 0, 1, 2, 3, 4, 5, 6, 7, 8, 9, 10, 11,
                                 12, 13, 14, 15);
}

// ---------------- Kernel 5: GEMM  out[b,c] = sc * sum_k Xq[b,k]*W[c,k] + bias[c]
// Block tile: 64 rows x 128 cols; 8 wave32 waves -> each wave 16x64 (4 WMMA D
// tiles, 2 WMMAs deep per k-stage). Double-buffered LDS, TDM DMA (wave0: W tile,
// wave1: X tile), 144B padded pitch for conflict-free ds_load fragments.
__global__ __launch_bounds__(256)
void k_gemm(const _Float16* __restrict__ Xq, const _Float16* __restrict__ Wq,
            const float* __restrict__ bias, const float* __restrict__ scalep,
            float* __restrict__ out, int Brows) {
  __shared__ __attribute__((aligned(128))) _Float16 Xs[2][64 * PITCH];
  __shared__ __attribute__((aligned(128))) _Float16 Ws[2][128 * PITCH];

  const int tid  = threadIdx.x;
  const int lane = tid & 31;
  const int wave = tid >> 5;
  const int rowBase = blockIdx.y * 64;
  const int colBase = blockIdx.x * 128;
  const int wRow = (wave & 3) * 16;   // wave's 16-row strip within tile
  const int wCol = (wave >> 2) * 64;  // wave's 64-col strip within tile

  // Fragment lane mapping: contiguous k order (k = 16*half + s), same for A and B
  const int half = lane >> 4;  // 0: K 0-15 (A & B); 1: K 16-31
  const int rl   = lane & 15;

  const int STEPS = CIN / KSTEP;
  const unsigned remH_x = (unsigned)(Brows > rowBase ? Brows - rowBase : 0);

  v8f acc[4] = {};

  // prologue: DMA k-stage 0 into buffer 0
  if (wave == 0) {
    tdm_load_tile_f16(lds_off(&Ws[0][0]), Wq + (size_t)colBase * CIN,
                      KSTEP, 128, (unsigned)CIN, (unsigned)(COUT - colBase),
                      (unsigned)CIN);
  } else if (wave == 1) {
    tdm_load_tile_f16(lds_off(&Xs[0][0]), Xq + (size_t)rowBase * CIN,
                      KSTEP, 64, (unsigned)CIN, remH_x, (unsigned)CIN);
  }
  if (wave < 2) __builtin_amdgcn_s_wait_tensorcnt(0);

#pragma unroll 1
  for (int s = 0; s < STEPS; ++s) {
    __syncthreads();                      // buf (s&1) published to all waves
    const int cur = s & 1;
    const int nxt = cur ^ 1;
    const bool pf = (s + 1 < STEPS);

    if (pf) {
      const int kk = (s + 1) * KSTEP;
      if (wave == 0) {
        tdm_load_tile_f16(lds_off(&Ws[nxt][0]),
                          Wq + (size_t)colBase * CIN + kk,
                          KSTEP, 128, (unsigned)(CIN - kk),
                          (unsigned)(COUT - colBase), (unsigned)CIN);
      } else if (wave == 1) {
        tdm_load_tile_f16(lds_off(&Xs[nxt][0]),
                          Xq + (size_t)rowBase * CIN + kk,
                          KSTEP, 64, (unsigned)(CIN - kk), remH_x,
                          (unsigned)CIN);
      }
    }

    // A fragments: K 0..31 and K 32..63 of this wave's 16 rows
    const _Float16* xrow = &Xs[cur][(wRow + rl) * PITCH];
    v16h a0 = cat8(*(const v8h*)(xrow + half * 16),
                   *(const v8h*)(xrow + half * 16 + 8));
    v16h a1 = cat8(*(const v8h*)(xrow + 32 + half * 16),
                   *(const v8h*)(xrow + 32 + half * 16 + 8));

    // Preload all B fragments (distinct regs -> overlapped dscnt waits)
    v16h b0[4], b1[4];
#pragma unroll
    for (int q = 0; q < 4; ++q) {
      const _Float16* wrow = &Ws[cur][(wCol + q * 16 + rl) * PITCH];
      b0[q] = cat8(*(const v8h*)(wrow + half * 16),
                   *(const v8h*)(wrow + half * 16 + 8));
      b1[q] = cat8(*(const v8h*)(wrow + 32 + half * 16),
                   *(const v8h*)(wrow + 32 + half * 16 + 8));
    }
#pragma unroll
    for (int q = 0; q < 4; ++q) {
      acc[q] = __builtin_amdgcn_wmma_f32_16x16x32_f16(
          false, a0, false, b0[q], (short)0, acc[q], false, false);
      acc[q] = __builtin_amdgcn_wmma_f32_16x16x32_f16(
          false, a1, false, b1[q], (short)0, acc[q], false, false);
    }

    if (pf && wave < 2) __builtin_amdgcn_s_wait_tensorcnt(0);
  }

  // Epilogue: D layout -> row = wRow + half*8 + v, col = wCol + q*16 + rl
  const float sc = scalep[0];
  if (rowBase + 64 <= Brows) {            // uniform fast path (B=512 case)
#pragma unroll
    for (int q = 0; q < 4; ++q) {
      const int col = colBase + wCol + q * 16 + rl;
      const float bv = bias[col];
      float* op = out + (size_t)(rowBase + wRow + half * 8) * COUT + col;
#pragma unroll
      for (int v = 0; v < 8; ++v) op[(size_t)v * COUT] = sc * acc[q][v] + bv;
    }
  } else {
#pragma unroll
    for (int q = 0; q < 4; ++q) {
      const int col = colBase + wCol + q * 16 + rl;
      const float bv = bias[col];
#pragma unroll
      for (int v = 0; v < 8; ++v) {
        const int row = rowBase + wRow + half * 8 + v;
        if (row < Brows) out[(size_t)row * COUT + col] = sc * acc[q][v] + bv;
      }
    }
  }
}

// ---------------- launcher ----------------
extern "C" void kernel_launch(void* const* d_in, const int* in_sizes, int n_in,
                              void* d_out, int out_size, void* d_ws,
                              size_t ws_size, hipStream_t stream) {
  const float* x      = (const float*)d_in[0];
  const float* Ysign  = (const float*)d_in[1];
  const float* Zsign  = (const float*)d_in[2];
  const float* Yscale = (const float*)d_in[3];
  const float* Zscale = (const float*)d_in[4];
  const float* A      = (const float*)d_in[5];
  const float* bias   = (const float*)d_in[6];
  float* out = (float*)d_out;

  const int n = in_sizes[0];        // B * 4096
  const int Brows = n / CIN;

  char* ws = (char*)d_ws;
  float*     pmax   = (float*)ws;                 // 256 f32
  float*     pmin   = (float*)(ws + 1024);        // 256 f32
  float*     scalep = (float*)(ws + 2048);        // 1 f32
  _Float16*  xq     = (_Float16*)(ws + 4096);     // n halves
  size_t xqBytes = ((size_t)n * 2 + 255) & ~(size_t)255;
  _Float16*  Wq     = (_Float16*)(ws + 4096 + xqBytes);  // 4096*4096 halves (32MB)

  k_minmax_partial<<<256, 256, 0, stream>>>(x, n, pmax, pmin);
  k_scale<<<1, 256, 0, stream>>>(pmax, pmin, scalep);
  k_quant<<<512, 256, 0, stream>>>(x, scalep, xq, n);
  k_buildW<<<JJ * KK, 256, 0, stream>>>(Ysign, Zsign, Yscale, Zscale, A, Wq);

  dim3 grid(COUT / 128, (Brows + 63) / 64);
  k_gemm<<<grid, 256, 0, stream>>>(xq, Wq, bias, scalep, out, Brows);
}
